// SpaceTimeOptimizedAttention_2362232013178
// MI455X (gfx1250) — matrix operand order repaired and checked by reading, not compiled, not run on hardware
//
#include <hip/hip_runtime.h>
#include <hip/hip_bf16.h>
#include <stdint.h>

// ---------------------------------------------------------------------------
// Types for CDNA5 WMMA (wave32): A/B = 16 x bf16 per lane, C/D = 8 x f32.
// ---------------------------------------------------------------------------
typedef __attribute__((ext_vector_type(16))) __bf16 v16bf;
typedef __attribute__((ext_vector_type(8)))  __bf16 bf16x8;
typedef __attribute__((ext_vector_type(4)))  __bf16 bf16x4;
typedef __attribute__((ext_vector_type(8)))  float  v8f;

#define NB   2
#define NS   2048
#define ND   2048
#define NH   16
#define NHD  128            // head dim
#define NM   (NB * NS)      // 4096 rows

__device__ __forceinline__ v16bf cat8(bf16x8 lo, bf16x8 hi) {
  return __builtin_shufflevector(lo, hi, 0,1,2,3,4,5,6,7,8,9,10,11,12,13,14,15);
}

__device__ __forceinline__ v8f wmma_bf16(v16bf a, v16bf b, v8f c) {
  // D = A(16x32) * B(32x16) + C(16x16), f32 accumulate
  return __builtin_amdgcn_wmma_f32_16x16x32_bf16(
      /*neg_a=*/false, a, /*neg_b=*/false, b,
      /*c_mod=*/(short)0, c, /*reuse_a=*/false, /*reuse_b=*/false);
}

// ---- CDNA5 async global->LDS copy (ASYNCcnt-tracked) ----------------------
__device__ __forceinline__ void async_copy_b128(uint32_t lds_off, const void* gaddr) {
  // GLOBAL_LOAD_ASYNC_TO_LDS_B128: vdst = LDS byte address, vaddr = 64-bit GVA
  asm volatile("global_load_async_to_lds_b128 %0, %1, off"
               :: "v"(lds_off), "v"(gaddr) : "memory");
}

__device__ __forceinline__ void wait_async0() {
#if __has_builtin(__builtin_amdgcn_s_wait_asynccnt)
  __builtin_amdgcn_s_wait_asynccnt(0);
#else
  asm volatile("s_wait_asynccnt 0x0" ::: "memory");
#endif
}

// Per-wave LDS fence (same-wave DS ops are in-order; hw wait + compiler fence)
#define WAVE_LDS_FENCE()                                   \
  do {                                                     \
    __builtin_amdgcn_wave_barrier();                       \
    asm volatile("s_wait_dscnt 0x0" ::: "memory");         \
  } while (0)

// ---------------------------------------------------------------------------
// f32 -> bf16 conversion (vectorized x4)
// ---------------------------------------------------------------------------
__global__ void cvt_f32_bf16(const float* __restrict__ src,
                             __bf16* __restrict__ dst, int n4) {
  int i = blockIdx.x * blockDim.x + threadIdx.x;
  if (i < n4) {
    float4 f = ((const float4*)src)[i];
    bf16x4 o = { (__bf16)f.x, (__bf16)f.y, (__bf16)f.z, (__bf16)f.w };
    ((bf16x4*)dst)[i] = o;
  }
}

// ---------------------------------------------------------------------------
// C[M,N] = A[M,K] @ Bw[N,K]^T      (A, Bw bf16 row-major; acc f32)
// Block: 256 threads = 8 waves; block tile 256(M) x 64(N); K-step 32.
// Each wave owns 32 M-rows (2 A-fragments, 8 accumulators).
// BOTH A and B tiles double-buffered in LDS via async global->LDS copies:
// 5 x b128 async per thread per K-step (16KB A + 4KB B), one counted
// async-wait + one barrier per step, zero synchronous global loads in the
// steady-state loop.
// ---------------------------------------------------------------------------
#define GTM 256
#define GTN 64
#define GKS 32

__global__ __launch_bounds__(256)
void gemm_bf16_nt(const __bf16* __restrict__ A, const __bf16* __restrict__ Bw,
                  __bf16* __restrict__ Cb, float* __restrict__ Cf,
                  int M, int N, int K, int outf32) {
  __shared__ __align__(16) __bf16 lA[2][GTM][GKS + 8];  // 2 x 256 x 40 bf16
  __shared__ __align__(16) __bf16 lB[2][GTN][GKS + 8];  // 2 x  64 x 40 bf16

  const int wave = threadIdx.x >> 5;
  const int lane = threadIdx.x & 31;
  const int half = lane >> 4;
  const int l16  = lane & 15;

  const int m0 = blockIdx.y * GTM;
  const int n0 = blockIdx.x * GTN;

  v8f acc[2][4];
#pragma unroll
  for (int mh = 0; mh < 2; ++mh)
#pragma unroll
    for (int nb = 0; nb < 4; ++nb) acc[mh][nb] = 0.0f;

  // cooperative async mapping: thread t owns 16B chunk (row tr, k tk)
  const int tr = threadIdx.x >> 2;          // 0..63
  const int tk = (threadIdx.x & 3) * 8;     // 0,8,16,24
  const __bf16* bsrc = Bw + (size_t)(n0 + tr) * K + tk;
  const __bf16* asrc[4];
#pragma unroll
  for (int c = 0; c < 4; ++c)
    asrc[c] = A + (size_t)(m0 + tr + c * 64) * K + tk;

  uint32_t ldsB[2], ldsA[2][4];
#pragma unroll
  for (int bu = 0; bu < 2; ++bu) {
    ldsB[bu] = (uint32_t)(uintptr_t)&lB[bu][tr][tk];
#pragma unroll
    for (int c = 0; c < 4; ++c)
      ldsA[bu][c] = (uint32_t)(uintptr_t)&lA[bu][tr + c * 64][tk];
  }

  // ---- prologue: kick off buffer 0 ----
  async_copy_b128(ldsB[0], bsrc);
#pragma unroll
  for (int c = 0; c < 4; ++c) async_copy_b128(ldsA[0][c], asrc[c]);

#pragma unroll 2
  for (int k0 = 0; k0 < K; k0 += GKS) {
    const int  cur  = (k0 >> 5) & 1;
    const bool more = (k0 + GKS) < K;

    // own async copies complete, then all waves rendezvous => whole tile valid
    wait_async0();
    __syncthreads();
    // other buffer's readers finished their ds_loads (dscnt-waited before
    // their wmmas) before the barrier -> safe to overwrite it now
    if (more) {
      async_copy_b128(ldsB[cur ^ 1], bsrc + k0 + GKS);
#pragma unroll
      for (int c = 0; c < 4; ++c)
        async_copy_b128(ldsA[cur ^ 1][c], asrc[c] + k0 + GKS);
      if (k0 + 2 * GKS < K) __builtin_prefetch(asrc[0] + k0 + 2 * GKS);
    }

    // ---- A fragments from LDS (A-layout: lanes 0-15 k0..7 & 16..23,
    //      lanes 16-31 k8..15 & 24..31) ----
    v16bf afrag[2];
#pragma unroll
    for (int mh = 0; mh < 2; ++mh) {
      const __bf16* ar = &lA[cur][wave * 32 + mh * 16 + l16][half * 8];
      afrag[mh] = cat8(*(const bf16x8*)ar, *(const bf16x8*)(ar + 16));
    }

    // ---- 8 WMMAs off the current LDS buffers ----
#pragma unroll
    for (int nb = 0; nb < 4; ++nb) {
      const __bf16* brow = &lB[cur][nb * 16 + l16][half * 8];
      v16bf bfrag = cat8(*(const bf16x8*)brow, *(const bf16x8*)(brow + 16));
#pragma unroll
      for (int mh = 0; mh < 2; ++mh)
        acc[mh][nb] = wmma_bf16(afrag[mh], bfrag, acc[mh][nb]);
    }
  }

  // C/D layout: lanes 0-15 -> row r, lanes 16-31 -> row r+8; col = lane%16
#pragma unroll
  for (int mh = 0; mh < 2; ++mh) {
    const int rowbase = m0 + wave * 32 + mh * 16 + half * 8;
#pragma unroll
    for (int nb = 0; nb < 4; ++nb) {
#pragma unroll
      for (int r = 0; r < 8; ++r) {
        const int row = rowbase + r;
        const int col = n0 + nb * 16 + l16;
        float v = acc[mh][nb][r];
        if (outf32) Cf[(size_t)row * N + col] = v;
        else        Cb[(size_t)row * N + col] = (__bf16)v;
      }
    }
  }
}

// ---------------------------------------------------------------------------
// Flash-style causal attention over bf16 Q,K,V stored as (B,S,D) with head h
// occupying columns [h*128, h*128+128). Block = 128 threads = 4 waves; each
// wave owns 16 query rows; streams keys 32 at a time with online softmax.
// grid = (S/64, B*H)
// ---------------------------------------------------------------------------
__global__ __launch_bounds__(128)
void attn_flash(const __bf16* __restrict__ Q, const __bf16* __restrict__ Km,
                const __bf16* __restrict__ V, __bf16* __restrict__ Ctx) {
  const float scale = 0.08838834764831845f;   // 1/sqrt(128)

  const int wave = threadIdx.x >> 5;
  const int lane = threadIdx.x & 31;
  const int half = lane >> 4;
  const int l16  = lane & 15;

  const int bh = blockIdx.y;
  const int b  = bh >> 4;
  const int h  = bh & 15;
  const int q0 = blockIdx.x * 64 + wave * 16;

  const size_t base = (size_t)b * NS * ND + (size_t)h * NHD;

  // Load Q fragments (16 rows x 128 cols -> 4 A-fragments of 16x32)
  v16bf qf[4];
  {
    const __bf16* qrow = Q + base + (size_t)(q0 + l16) * ND + half * 8;
#pragma unroll
    for (int c = 0; c < 4; ++c)
      qf[c] = cat8(*(const bf16x8*)(qrow + c * 32),
                   *(const bf16x8*)(qrow + c * 32 + 16));
  }

  v8f ctx[8];
  float mi[8], li[8];
#pragma unroll
  for (int i = 0; i < 8; ++i) { ctx[i] = 0.0f; mi[i] = -__builtin_inff(); li[i] = 0.0f; }

  __shared__ __align__(16) __bf16 psh[4][16][40];   // per-wave 16x32 P tile

  for (int j0 = 0; j0 <= q0 + 15; j0 += 32) {
    // keep the next key/value block warm in the WGP cache
    if (j0 + 32 <= q0 + 15) {
      __builtin_prefetch(Km + base + (size_t)(j0 + 32 + l16) * ND);
      __builtin_prefetch(V  + base + (size_t)(j0 + 32 + lane) * ND);
    }

    // ---- S = Q K^T (16 x 32), two 16x16 C-tiles ----
    v8f sc[2];
#pragma unroll
    for (int c = 0; c < 2; ++c) {
      sc[c] = 0.0f;
      const __bf16* krow = Km + base + (size_t)(j0 + c * 16 + l16) * ND + half * 8;
#pragma unroll
      for (int kc = 0; kc < 4; ++kc) {
        v16bf kf = cat8(*(const bf16x8*)(krow + kc * 32),
                        *(const bf16x8*)(krow + kc * 32 + 16));
        sc[c] = wmma_bf16(qf[kc], kf, sc[c]);
      }
    }

    // ---- mask + online softmax (row stats reduced over 16-lane halves) ----
    float alpha[8];
#pragma unroll
    for (int r = 0; r < 8; ++r) {
      const int qi = q0 + half * 8 + r;
      float s0 = sc[0][r] * scale; if (j0 + l16      > qi) s0 = -__builtin_inff();
      float s1 = sc[1][r] * scale; if (j0 + 16 + l16 > qi) s1 = -__builtin_inff();

      float mx = fmaxf(s0, s1);
#pragma unroll
      for (int d = 1; d < 16; d <<= 1) mx = fmaxf(mx, __shfl_xor(mx, d, 32));
      const float mnew = fmaxf(mi[r], mx);
      const float a = __expf(mi[r] - mnew);
      const float p0 = __expf(s0 - mnew);
      const float p1 = __expf(s1 - mnew);
      float ps = p0 + p1;
#pragma unroll
      for (int d = 1; d < 16; d <<= 1) ps += __shfl_xor(ps, d, 32);
      li[r] = li[r] * a + ps;
      mi[r] = mnew;
      alpha[r] = a;

      const int m = half ? (r + 8) : r;        // C-layout row for this lane
      psh[wave][m][l16]      = (__bf16)p0;
      psh[wave][m][16 + l16] = (__bf16)p1;
    }
    WAVE_LDS_FENCE();

    // ---- P back as A-fragment (16x32) ----
    const __bf16* prow = &psh[wave][l16][half * 8];
    v16bf pf = cat8(*(const bf16x8*)prow, *(const bf16x8*)(prow + 16));

    // ---- rescale running context, then ctx += P @ V ----
#pragma unroll
    for (int nb = 0; nb < 8; ++nb) {
#pragma unroll
      for (int r = 0; r < 8; ++r) ctx[nb][r] *= alpha[r];
    }
#pragma unroll
    for (int nb = 0; nb < 8; ++nb) {
      const __bf16* vrow = V + base + (size_t)(j0 + half * 8) * ND + nb * 16 + l16;
      v16bf vf;
#pragma unroll
      for (int s = 0; s < 16; ++s) vf[s] = vrow[(size_t)(s < 8 ? s : s + 8) * ND];
      ctx[nb] = wmma_bf16(pf, vf, ctx[nb]);
    }
    WAVE_LDS_FENCE();   // before next iteration overwrites psh
  }

  // ---- normalize and write context (bf16) into (B,S,D) layout ----
#pragma unroll
  for (int nb = 0; nb < 8; ++nb) {
#pragma unroll
    for (int r = 0; r < 8; ++r) {
      const int row = q0 + half * 8 + r;
      const float v = ctx[nb][r] / li[r];
      Ctx[base + (size_t)row * ND + nb * 16 + l16] = (__bf16)v;
    }
  }
}

// ---------------------------------------------------------------------------
// Host-side orchestration
// ---------------------------------------------------------------------------
extern "C" void kernel_launch(void* const* d_in, const int* in_sizes, int n_in,
                              void* d_out, int out_size, void* d_ws, size_t ws_size,
                              hipStream_t stream) {
  const float* x  = (const float*)d_in[0];
  const float* Wq = (const float*)d_in[1];
  const float* Wk = (const float*)d_in[2];
  const float* Wv = (const float*)d_in[3];
  const float* Wo = (const float*)d_in[4];
  float* out = (float*)d_out;

  const size_t MB = 1024ull * 1024ull;
  uint8_t* w = (uint8_t*)d_ws;
  __bf16* xb  = (__bf16*)(w + 0 * MB);     // 16 MB  (4096x2048)
  __bf16* wqb = (__bf16*)(w + 16 * MB);    // 8 MB
  __bf16* wkb = (__bf16*)(w + 24 * MB);
  __bf16* wvb = (__bf16*)(w + 32 * MB);
  __bf16* wob = (__bf16*)(w + 40 * MB);
  __bf16* qb  = (__bf16*)(w + 48 * MB);    // 16 MB each
  __bf16* kb  = (__bf16*)(w + 64 * MB);
  __bf16* vb  = (__bf16*)(w + 80 * MB);
  __bf16* cxb = (__bf16*)(w + 96 * MB);    // total 112 MB

  const int nx4 = (NM * ND) / 4;       // 2,097,152
  const int nw4 = (ND * ND) / 4;       // 1,048,576
  cvt_f32_bf16<<<(nx4 + 255) / 256, 256, 0, stream>>>(x,  xb,  nx4);
  cvt_f32_bf16<<<(nw4 + 255) / 256, 256, 0, stream>>>(Wq, wqb, nw4);
  cvt_f32_bf16<<<(nw4 + 255) / 256, 256, 0, stream>>>(Wk, wkb, nw4);
  cvt_f32_bf16<<<(nw4 + 255) / 256, 256, 0, stream>>>(Wv, wvb, nw4);
  cvt_f32_bf16<<<(nw4 + 255) / 256, 256, 0, stream>>>(Wo, wob, nw4);

  dim3 ggrid(ND / GTN, NM / GTM);      // (32, 16)
  gemm_bf16_nt<<<ggrid, 256, 0, stream>>>(xb, wqb, qb, nullptr, NM, ND, ND, 0);
  gemm_bf16_nt<<<ggrid, 256, 0, stream>>>(xb, wkb, kb, nullptr, NM, ND, ND, 0);
  gemm_bf16_nt<<<ggrid, 256, 0, stream>>>(xb, wvb, vb, nullptr, NM, ND, ND, 0);

  dim3 agrid(NS / 64, NB * NH);        // (32, 32)
  attn_flash<<<agrid, 128, 0, stream>>>(qb, kb, vb, cxb);

  gemm_bf16_nt<<<ggrid, 256, 0, stream>>>(cxb, wob, nullptr, out, NM, ND, ND, 1);
}
